// EdgeAwareCrossAttention_88003879895613
// MI455X (gfx1250) — matrix operand order repaired and checked by reading, not compiled, not run on hardware
//
#include <hip/hip_runtime.h>

#define BATCH 4
#define CH    256
#define NI    32
#define NN    4096
#define NBLK  (NN / 32)   // 128 key blocks
#define VPAD  40          // 32 bf16 payload + 8 pad -> 80B rows, conflict-free b128 reads

typedef __attribute__((ext_vector_type(16))) __bf16 v16bf;
typedef __attribute__((ext_vector_type(8)))  float  v8f;

union BF16x16 { v16bf v; uint4 q[2]; };

// ---------------------------------------------------------------------------
// 1x1-conv projection: out[o, n] = sum_c W[o,c] * in[b,c,n] + bias[o]
// OMODE 0: out layout [B][N][Ototal] bf16 (Q/K rows, contiguous 32 per token)
// OMODE 1: out layout [B][Ototal][N] bf16 (V, channel-major rows)
// ---------------------------------------------------------------------------
template <int OMODE>
__global__ __launch_bounds__(256) void proj_kernel(
    const float* __restrict__ W, const float* __restrict__ bias,
    const float* __restrict__ in, __bf16* __restrict__ outp, int Ototal)
{
    __shared__ __align__(16) float swt[CH * 32];  // transposed weight tile [c][j]

    const int tid = threadIdx.x;
    const int o0  = blockIdx.y * 32;

    for (int idx = tid; idx < CH * 32; idx += 256) {
        int c = idx >> 5, j = idx & 31;
        swt[idx] = W[(size_t)(o0 + j) * CH + c];
    }
    __syncthreads();

    const int b  = blockIdx.x / (NN / 256);
    const int n  = (blockIdx.x % (NN / 256)) * 256 + tid;
    const float* ip = in + (size_t)b * CH * NN + n;

    float acc[32];
#pragma unroll
    for (int j = 0; j < 32; ++j) acc[j] = bias[o0 + j];

    for (int c = 0; c < CH; ++c) {
        float f = ip[(size_t)c * NN];
        const float4* w4 = (const float4*)&swt[c * 32];
#pragma unroll
        for (int jj = 0; jj < 8; ++jj) {
            float4 w = w4[jj];
            acc[4 * jj + 0] += w.x * f;
            acc[4 * jj + 1] += w.y * f;
            acc[4 * jj + 2] += w.z * f;
            acc[4 * jj + 3] += w.w * f;
        }
    }

    if (OMODE == 0) {
        __bf16* op = outp + ((size_t)b * NN + n) * Ototal + o0;
#pragma unroll
        for (int j = 0; j < 32; ++j) op[j] = (__bf16)acc[j];
    } else {
#pragma unroll
        for (int j = 0; j < 32; ++j)
            outp[((size_t)b * Ototal + o0 + j) * NN + n] = (__bf16)acc[j];
    }
}

// ---------------------------------------------------------------------------
// Flash cross-attention with async-LDS double-buffered K/V staging.
// 4 waves / block; wave owns 16 query rows, full C=256 output in 16 f32 tiles.
// ---------------------------------------------------------------------------
__device__ __forceinline__ void async_b128(unsigned lds_off, const void* gptr) {
    unsigned long long ga = (unsigned long long)(uintptr_t)gptr;
    asm volatile("global_load_async_to_lds_b128 %0, %1, off"
                 :: "v"(lds_off), "v"(ga) : "memory");
}
__device__ __forceinline__ void wait_async0() {
    asm volatile("s_wait_asynccnt 0x0" ::: "memory");
}

__global__ __launch_bounds__(128) void attn_kernel(
    const __bf16* __restrict__ Qb,   // [B][N][32]
    const __bf16* __restrict__ Kb,   // [B][N][32]
    const __bf16* __restrict__ Vb,   // [B][C][N]
    const float*  __restrict__ feat, // [B][C][N] residual
    float*        __restrict__ out)  // [B][C][N]
{
    __shared__ __align__(16) __bf16 sV[2][CH][VPAD]; // V key-block, 80B rows
    __shared__ __align__(16) __bf16 sK[2][32][VPAD]; // K key-block
    __shared__ __align__(16) __bf16 pP[4][16][VPAD]; // per-wave P tile [q][key]
    __shared__ float rstat[4][16];                   // per-wave row stats

    const int tid  = threadIdx.x;
    const int wave = tid >> 5;
    const int lane = tid & 31;
    const int lm   = lane & 15;   // N-col / row-id within 16
    const int hi   = lane >> 4;   // half-wave select
    const int kloA = hi ? 16 : 0; // A-matrix K chunk:  [klo,klo+16)
    const int kloB = hi ? 16 : 0; // B-matrix K chunk:  [klo,klo+16)

    const int b  = blockIdx.x / (NN / 64);
    const int q0 = (blockIdx.x % (NN / 64)) * 64 + wave * 16;

    const __bf16* Vbase = Vb + (size_t)b * CH * NN;
    const __bf16* Kbase = Kb + (size_t)b * NN * NI;

    // cooperative async fill of one 32-key block (V: 1024 x 16B, K: 128 x 16B)
    auto fill = [&](int buf, int m0) {
#pragma unroll
        for (int i = 0; i < 8; ++i) {
            int idx = tid + i * 128;          // 0..1023
            int c = idx >> 2, j = idx & 3;
            async_b128((unsigned)(uintptr_t)&sV[buf][c][j * 8],
                       Vbase + (size_t)c * NN + m0 + j * 8);
        }
        {
            int c = tid >> 2, j = tid & 3;    // key row 0..31, chunk 0..3
            async_b128((unsigned)(uintptr_t)&sK[buf][c][j * 8],
                       Kbase + (size_t)(m0 + c) * NI + j * 8);
        }
    };

    // Q A-tile (16 q-rows x 32 inter), reused across all key blocks
    BF16x16 qa;
    {
        const __bf16* qrow = Qb + ((size_t)b * NN + q0 + lm) * NI + kloA;
        qa.q[0] = *(const uint4*)(qrow);
        qa.q[1] = *(const uint4*)(qrow + 8);
    }

    const v8f zero = {};
    v8f O[16];
#pragma unroll
    for (int t = 0; t < 16; ++t) O[t] = zero;
    float m_i[8], l_i[8];
#pragma unroll
    for (int r = 0; r < 8; ++r) { m_i[r] = -1e30f; l_i[r] = 0.f; }

    fill(0, 0);
    wait_async0();
    __syncthreads();

    for (int it = 0; it < NBLK; ++it) {
        const int m0  = it * 32;
        const int buf = it & 1;
        // prefetch next key block into the other buffer (wraps harmlessly)
        fill(buf ^ 1, (it + 1 < NBLK) ? m0 + 32 : 0);

        // ---- S = Q x K^T for 32 keys (two 16x16x32 bf16 WMMAs) ----
        BF16x16 kb0, kb1;
        kb0.q[0] = *(const uint4*)(&sK[buf][lm][kloB]);
        kb0.q[1] = *(const uint4*)(&sK[buf][lm][kloB + 8]);
        kb1.q[0] = *(const uint4*)(&sK[buf][lm + 16][kloB]);
        kb1.q[1] = *(const uint4*)(&sK[buf][lm + 16][kloB + 8]);
        v8f S0 = __builtin_amdgcn_wmma_f32_16x16x32_bf16(false, qa.v, false, kb0.v,
                                                         (short)0, zero, false, false);
        v8f S1 = __builtin_amdgcn_wmma_f32_16x16x32_bf16(false, qa.v, false, kb1.v,
                                                         (short)0, zero, false, false);

        // ---- online softmax (row = q, spread over a 16-lane half) ----
        float fac[8];
#pragma unroll
        for (int r = 0; r < 8; ++r) {
            float v = fmaxf(S0[r], S1[r]);
            v = fmaxf(v, __shfl_xor(v, 1));
            v = fmaxf(v, __shfl_xor(v, 2));
            v = fmaxf(v, __shfl_xor(v, 4));
            v = fmaxf(v, __shfl_xor(v, 8));
            float mn = fmaxf(m_i[r], v);
            fac[r]   = __expf(m_i[r] - mn);
            m_i[r]   = mn;
            float p0 = __expf(S0[r] - mn);
            float p1 = __expf(S1[r] - mn);
            l_i[r]   = l_i[r] * fac[r] + p0 + p1;
            S0[r] = p0; S1[r] = p1;
        }

        // ---- stash P (bf16) + rescale factor into per-wave LDS ----
#pragma unroll
        for (int r = 0; r < 8; ++r) {
            pP[wave][r + 8 * hi][lm]      = (__bf16)S0[r];
            pP[wave][r + 8 * hi][lm + 16] = (__bf16)S1[r];
        }
        if (lm == 0) {
#pragma unroll
            for (int r = 0; r < 8; ++r) rstat[wave][r + 8 * hi] = fac[r];
        }
        __builtin_amdgcn_wave_barrier();

        const float facq = rstat[wave][lm];  // per-column (=q) rescale for O^T
        BF16x16 pb;                          // P^T as B-matrix (32 keys x 16 q)
        pb.q[0] = *(const uint4*)(&pP[wave][lm][kloB]);
        pb.q[1] = *(const uint4*)(&pP[wave][lm][kloB + 8]);
        __builtin_amdgcn_wave_barrier();

        // ---- rescale only when some row's max actually moved (wave-uniform) ----
        if (__any(facq != 1.0f)) {
#pragma unroll
            for (int t = 0; t < 16; ++t)
#pragma unroll
                for (int r = 0; r < 8; ++r) O[t][r] *= facq;
        }

        // ---- O^T[c,q] += V[c,keys] x P^T[keys,q], 2-deep pipelined V tiles ----
        BF16x16 va[2];
        va[0].q[0] = *(const uint4*)(&sV[buf][lm][kloA]);
        va[0].q[1] = *(const uint4*)(&sV[buf][lm][kloA + 8]);
#pragma unroll
        for (int t = 0; t < 16; ++t) {
            const int cur = t & 1;
            if (t < 15) {
                va[cur ^ 1].q[0] = *(const uint4*)(&sV[buf][(t + 1) * 16 + lm][kloA]);
                va[cur ^ 1].q[1] = *(const uint4*)(&sV[buf][(t + 1) * 16 + lm][kloA + 8]);
            }
            O[t] = __builtin_amdgcn_wmma_f32_16x16x32_bf16(false, va[cur].v, false, pb.v,
                                                           (short)0, O[t], false, false);
        }

        wait_async0();     // our fills for buf^1 are done
        __syncthreads();   // everyone done reading buf + filling buf^1
    }

    // ---- epilogue: finish row sums, normalize, residual add, store ----
#pragma unroll
    for (int r = 0; r < 8; ++r) {
        float v = l_i[r];
        v += __shfl_xor(v, 1);
        v += __shfl_xor(v, 2);
        v += __shfl_xor(v, 4);
        v += __shfl_xor(v, 8);
        l_i[r] = v;
    }
    if (lm == 0) {
#pragma unroll
        for (int r = 0; r < 8; ++r) rstat[wave][r + 8 * hi] = l_i[r];
    }
    __builtin_amdgcn_wave_barrier();
    const float linv = 1.0f / rstat[wave][lm];

    const size_t nq = (size_t)q0 + lm;
#pragma unroll
    for (int t = 0; t < 16; ++t) {
#pragma unroll
        for (int r = 0; r < 8; ++r) {
            int c = t * 16 + r + 8 * hi;
            size_t idx = ((size_t)b * CH + c) * NN + nq;
            out[idx] = O[t][r] * linv + feat[idx];
        }
    }
}

// ---------------------------------------------------------------------------
extern "C" void kernel_launch(void* const* d_in, const int* in_sizes, int n_in,
                              void* d_out, int out_size, void* d_ws, size_t ws_size,
                              hipStream_t stream) {
    const float* feat = (const float*)d_in[0];
    const float* edge = (const float*)d_in[1];
    const float* wq   = (const float*)d_in[2];
    const float* bq   = (const float*)d_in[3];
    const float* wk   = (const float*)d_in[4];
    const float* bk   = (const float*)d_in[5];
    const float* wv   = (const float*)d_in[6];
    const float* bv   = (const float*)d_in[7];
    float* out = (float*)d_out;

    __bf16* Qb = (__bf16*)d_ws;                    // [B][N][32]   1 MB
    __bf16* Kb = Qb + (size_t)BATCH * NN * NI;     // [B][N][32]   1 MB
    __bf16* Vb = Kb + (size_t)BATCH * NN * NI;     // [B][C][N]    8 MB

    dim3 pblk(256);
    proj_kernel<0><<<dim3(BATCH * (NN / 256), 1),       pblk, 0, stream>>>(wq, bq, feat, Qb, NI);
    proj_kernel<0><<<dim3(BATCH * (NN / 256), 1),       pblk, 0, stream>>>(wk, bk, edge, Kb, NI);
    proj_kernel<1><<<dim3(BATCH * (NN / 256), CH / 32), pblk, 0, stream>>>(wv, bv, edge, Vb, CH);

    attn_kernel<<<dim3(BATCH * (NN / 64)), dim3(128), 0, stream>>>(Qb, Kb, Vb, feat, out);
}
